// GaussianRenderer_36670430773891
// MI455X (gfx1250) — matrix operand order repaired and checked by reading, not compiled, not run on hardware
//
#include <hip/hip_runtime.h>

typedef __attribute__((ext_vector_type(16))) _Float16 v16h;
typedef __attribute__((ext_vector_type(8)))  _Float16 v8h;
typedef __attribute__((ext_vector_type(8)))  float    v8f;

#define NUM_TILE 16
#define TILE     32
#define KPT      256
#define W_IMG    (NUM_TILE * TILE)   // 512
#define WROW     40                  // padded W row stride in halfs (80B: 16B aligned, conflict-free b128)

__global__ __launch_bounds__(1024, 1)
void gsplat_render(const float* __restrict__ pos2d,
                   const float* __restrict__ cov2d,
                   const float* __restrict__ opacity,
                   const float* __restrict__ color,
                   const int*   __restrict__ tile_indices,
                   const int*   __restrict__ tile_counts,
                   float* __restrict__ out)
{
    __shared__ __align__(16) float4   s_g0[KPT];        // px, py, ia, (ib+ic)
    __shared__ __align__(16) float2   s_g1[KPT];        // idd, opacity
    __shared__ __align__(16) _Float16 s_colT[4 * KPT];  // channel-major: [c][k], c==3 row is zero pad
    __shared__ __align__(16) _Float16 s_W[1024 * WROW]; // per-chunk weight matrix (f16), wave-private rows

    const int t    = blockIdx.x;
    const int tx   = t / NUM_TILE;
    const int ty   = t % NUM_TILE;
    const int p    = threadIdx.x;     // pixel id within tile: p = x*32 + y
    const int lane = p & 31;
    const int wave = p >> 5;          // == pixel x (tile row)

    // CDNA5 prefetch of this tile's index list (global_prefetch_b8)
    __builtin_prefetch(&tile_indices[t * KPT + (p & (KPT - 1))], 0, 1);

    // ---- Phase 1: gather + preprocess one gaussian per thread (threads 0..255) ----
    if (p < KPT) {
        const int gi = tile_indices[t * KPT + p];
        const float a  = cov2d[gi * 4 + 0];
        const float b  = cov2d[gi * 4 + 1];
        const float c2 = cov2d[gi * 4 + 2];
        const float d  = cov2d[gi * 4 + 3];
        const float inv = 1.0f / (a * d - b * c2);
        s_g0[p] = make_float4(pos2d[gi * 2 + 0], pos2d[gi * 2 + 1], d * inv, -(b + c2) * inv);
        s_g1[p] = make_float2(a * inv, opacity[gi]);
        s_colT[0 * KPT + p] = (_Float16)color[gi * 3 + 0];
        s_colT[1 * KPT + p] = (_Float16)color[gi * 3 + 1];
        s_colT[2 * KPT + p] = (_Float16)color[gi * 3 + 2];
        s_colT[3 * KPT + p] = (_Float16)0.0f;
    }
    const int count = tile_counts[t];
    __syncthreads();   // only barrier needed: shared params ready

    // pixel center in world coords (x = wave, y = lane)
    const float fx = (float)(tx * TILE + wave) + 0.5f;
    const float fy = (float)(ty * TILE + lane) + 0.5f;

    float trans = 1.0f;               // running transmittance, sequential over k
    v8f acc0 = {};                    // WMMA C/D accumulators: pixels 32w..32w+15 (x16 channels)
    v8f acc1 = {};                    // pixels 32w+16..32w+31

    // Constant fragment addressing (WMMA 16-bit A 16x32 / B 32x16 lane layouts).
    // NOTE: rowA0/rowA1 are within [32*wave, 32*wave+32) -> s_W traffic is wave-private,
    // so no workgroup barriers are required around the W produce/consume phases.
    const int rowA0  = wave * 32 + (lane & 15);
    const int rowA1  = rowA0 + 16;
    const int koffA  = (lane < 16) ? 0 : 8;    // halfs: lanes 0-15 hold K 0-7/16-23, lanes 16-31 hold K 8-15/24-31
    const int cN     = lane & 15;              // B column = output channel
    const int cidx   = (cN < 3) ? cN : 3;      // channel >= 3 reads the zero row (branchless, EXEC stays full)
    const int kHalfB = (lane < 16) ? 0 : 8;    // B: lanes 0-15 K 0-7/16-23, lanes 16-31 K 8-15/24-31

    _Float16* wr = &s_W[p * WROW];

    #pragma unroll 1
    for (int ch = 0; ch < KPT / 32; ++ch) {
        const int kbase = ch * 32;

        // ---- Phase 2: sequential alpha compositing, batched b128 stores of w ----
        #pragma unroll 1
        for (int kq = 0; kq < 4; ++kq) {
            v8h wbuf;
            #pragma unroll
            for (int j = 0; j < 8; ++j) {
                const int k = kbase + kq * 8 + j;
                const float4 g0 = s_g0[k];
                const float2 g1 = s_g1[k];
                const float dx   = fx - g0.x;
                const float dy   = fy - g0.y;
                const float maha = dx * dx * g0.z + dx * dy * g0.w + dy * dy * g1.x;
                float alpha = g1.y * __expf(-0.5f * maha);
                alpha = fminf(fmaxf(alpha, 0.01f), 0.99f);
                alpha = (k < count) ? alpha : 0.0f;
                wbuf[j] = (_Float16)(alpha * trans);
                trans *= (1.0f - alpha);
            }
            *(v8h*)(wr + kq * 8) = wbuf;
        }

        // ---- Phase 3: matrix-pipe color accumulation  D += W_chunk x colors ----
        // A fragments: this wave's two 16-pixel M-tiles (wave-private rows of s_W)
        const _Float16* r0 = &s_W[rowA0 * WROW + koffA];
        const _Float16* r1 = &s_W[rowA1 * WROW + koffA];
        const v8h a0lo = *(const v8h*)(r0);
        const v8h a0hi = *(const v8h*)(r0 + 16);
        const v8h a1lo = *(const v8h*)(r1);
        const v8h a1hi = *(const v8h*)(r1 + 16);
        const v16h A0 = __builtin_shufflevector(a0lo, a0hi, 0,1,2,3,4,5,6,7,8,9,10,11,12,13,14,15);
        const v16h A1 = __builtin_shufflevector(a1lo, a1hi, 0,1,2,3,4,5,6,7,8,9,10,11,12,13,14,15);

        // B fragment (32x16): elements 0-7 = colT[channel][kbase + kHalfB + i], 8-15 = +16 (matches A)
        const _Float16* bp = &s_colT[cidx * KPT + kbase + kHalfB];
        const v8h blo = *(const v8h*)(bp);
        const v8h bhi = *(const v8h*)(bp + 16);
        const v16h B = __builtin_shufflevector(blo, bhi, 0,1,2,3,4,5,6,7,8,9,10,11,12,13,14,15);

        acc0 = __builtin_amdgcn_wmma_f32_16x16x32_f16(false, A0, false, B,
                                                      (short)0, acc0, false, false);
        acc1 = __builtin_amdgcn_wmma_f32_16x16x32_f16(false, A1, false, B,
                                                      (short)0, acc1, false, false);
    }

    // ---- Phase 4: writeout.  C/D layout: reg r, lane -> M = r + 8*(lane>=16), N = lane%16 ----
    if (cN < 3) {
        float* orow = out + ((size_t)(tx * TILE + wave) * W_IMG + (size_t)(ty * TILE)) * 3 + cN;
        #pragma unroll
        for (int r = 0; r < 8; ++r) {
            const int y0 = r + ((lane >> 4) << 3);
            orow[y0 * 3]        = acc0[r];   // pixels y = 0..15
            orow[(y0 + 16) * 3] = acc1[r];   // pixels y = 16..31
        }
    }
}

extern "C" void kernel_launch(void* const* d_in, const int* in_sizes, int n_in,
                              void* d_out, int out_size, void* d_ws, size_t ws_size,
                              hipStream_t stream) {
    (void)in_sizes; (void)n_in; (void)out_size; (void)d_ws; (void)ws_size;
    gsplat_render<<<dim3(NUM_TILE * NUM_TILE), dim3(1024), 0, stream>>>(
        (const float*)d_in[0],   // pos2d
        (const float*)d_in[1],   // cov2d
        (const float*)d_in[2],   // opacity
        (const float*)d_in[3],   // color
        (const int*)d_in[4],     // tile_indices
        (const int*)d_in[5],     // tile_counts
        (float*)d_out);
}
